// ProteinMPNN_19146964206157
// MI455X (gfx1250) — compile-verified, matrix operand order checked by reading, not run on hardware
//
#include <hip/hip_runtime.h>

#define NB 4
#define NL 1536
#define NK 30
#define NM (NB*NL*NK)      // 184320 edges, divisible by 16
#define NED 128            // edge feature out dim
#define KD  192            // 167 padded to 6 * 32

typedef __attribute__((ext_vector_type(16))) _Float16 v16h;
typedef __attribute__((ext_vector_type(8)))  float    v8f;
typedef __attribute__((ext_vector_type(4)))  float    v4f;

union HFrag { v16h h; v4f f[2]; };

__device__ __forceinline__ float dot3(const float* a, const float* b) {
  return a[0]*b[0] + a[1]*b[1] + a[2]*b[2];
}
__device__ __forceinline__ void cross3(const float* a, const float* b, float* c) {
  c[0] = a[1]*b[2] - a[2]*b[1];
  c[1] = a[2]*b[0] - a[0]*b[2];
  c[2] = a[0]*b[1] - a[1]*b[0];
}
__device__ __forceinline__ void norml3(const float* a, float* o) {
  float r = rsqrtf(dot3(a,a) + 1e-12f);
  o[0]=a[0]*r; o[1]=a[1]*r; o[2]=a[2]*r;
}
__device__ __forceinline__ float sgnf(float x){ return (x>0.f)?1.f:((x<0.f)?-1.f:0.f); }

// ---------------------------------------------------------------------------
// Kernel 1: per-node orientation frame O (3x3 rows: o1, n2, o1 x n2)
// Valid for l in [1, L-3]; zeros elsewhere (matches jnp.pad (1,2)).
// ---------------------------------------------------------------------------
__global__ void k_frames(const float* __restrict__ Ca, float* __restrict__ Ofr) {
  int idx = blockIdx.x * blockDim.x + threadIdx.x;
  if (idx >= NB*NL) return;
  int l = idx % NL;
  float o[9] = {0,0,0, 0,0,0, 0,0,0};
  if (l >= 1 && l <= NL-3) {
    const float* base = Ca + (size_t)(idx - l) * 3;   // chain start
    float u[2][3];
    #pragma unroll
    for (int s = 0; s < 2; ++s) {
      int t = l - 1 + s;
      float dX[3] = { base[(t+1)*3+0]-base[t*3+0],
                      base[(t+1)*3+1]-base[t*3+1],
                      base[(t+1)*3+2]-base[t*3+2] };
      float dn = sqrtf(dot3(dX,dX) + 1e-12f);
      float band = (dn > 3.6f && dn < 4.0f) ? 1.f : 0.f;
      float v[3] = { dX[0]*band, dX[1]*band, dX[2]*band };
      norml3(v, u[s]);
    }
    float diff[3] = { u[0][0]-u[1][0], u[0][1]-u[1][1], u[0][2]-u[1][2] };
    float o1[3]; norml3(diff, o1);
    float cr[3]; cross3(u[0], u[1], cr);
    float n2[3]; norml3(cr, n2);
    float c2[3]; cross3(o1, n2, c2);
    o[0]=o1[0]; o[1]=o1[1]; o[2]=o1[2];
    o[3]=n2[0]; o[4]=n2[1]; o[5]=n2[2];
    o[6]=c2[0]; o[7]=c2[1]; o[8]=c2[2];
  }
  float* dst = Ofr + (size_t)idx * 9;
  #pragma unroll
  for (int s = 0; s < 9; ++s) dst[s] = o[s];
}

// ---------------------------------------------------------------------------
// Kernel 2: W_edge f32[128][167] -> f16[128][192] (zero padded), n-major.
// ---------------------------------------------------------------------------
__global__ void k_wcvt(const float* __restrict__ We, _Float16* __restrict__ Wh) {
  int idx = blockIdx.x * blockDim.x + threadIdx.x;
  if (idx >= NED * KD) return;
  int n = idx / KD, k = idx % KD;
  float v = (k < 167) ? We[n*167 + k] : 0.f;
  Wh[idx] = (_Float16)v;
}

// ---------------------------------------------------------------------------
// Kernel 3: masked distances + top-K (ascending Dadj, tie -> lower index).
// One 256-thread block per (b, i) row. Ca chain (18 KB) is staged into LDS
// with gfx1250 async-to-LDS loads (ASYNCcnt), then the distance loop runs
// entirely out of LDS. Also writes the E_idx output tail.
// ---------------------------------------------------------------------------
__global__ __launch_bounds__(256) void k_topk(const float* __restrict__ Ca,
                                              const float* __restrict__ mask,
                                              int*   __restrict__ Eidx,
                                              float* __restrict__ Dnei,
                                              float* __restrict__ IdxOut) {
  __shared__ alignas(16) float CaLds[NL*3];    // 18 KB staged chain
  __shared__ float Drow[NL];
  __shared__ float redV[256];
  __shared__ int   redI[256];

  int bi = blockIdx.x;
  int b  = bi / NL;
  int i  = bi % NL;
  int t  = threadIdx.x;
  const float* CaB = Ca   + (size_t)b * NL * 3;
  const float* mB  = mask + (size_t)b * NL;

  // ---- async stage Ca chain into LDS (1152 x 16B chunks) ----
  for (int c = t; c < (NL*3)/4; c += 256) {
    unsigned lds  = (unsigned)(uintptr_t)(&CaLds[c*4]);
    unsigned voff = (unsigned)c * 16u;
    asm volatile("global_load_async_to_lds_b128 %0, %1, %2"
                 :: "v"(lds), "v"(voff), "s"(CaB) : "memory");
  }
  asm volatile("s_wait_asynccnt 0" ::: "memory");
  __syncthreads();

  float ci0 = CaLds[i*3+0], ci1 = CaLds[i*3+1], ci2 = CaLds[i*3+2];
  float mi  = mB[i];

  float lmax = 0.f;
  for (int j = t; j < NL; j += 256) {
    float dx = ci0 - CaLds[j*3+0];
    float dy = ci1 - CaLds[j*3+1];
    float dz = ci2 - CaLds[j*3+2];
    float d  = sqrtf(dx*dx + dy*dy + dz*dz + 1e-6f);
    float D  = mi * mB[j] * d;
    Drow[j] = D;
    lmax = fmaxf(lmax, D);
  }
  redV[t] = lmax;
  __syncthreads();
  for (int s = 128; s > 0; s >>= 1) {
    if (t < s) redV[t] = fmaxf(redV[t], redV[t+s]);
    __syncthreads();
  }
  float Dmax = redV[0];
  __syncthreads();
  for (int j = t; j < NL; j += 256) {
    float m2 = mi * mB[j];
    Drow[j] += (1.f - m2) * Dmax;
  }
  __syncthreads();

  for (int k = 0; k < NK; ++k) {
    float bv = 3.4e38f; int bj = -1;
    for (int j = t; j < NL; j += 256) {
      float v = Drow[j];
      if (v < bv) { bv = v; bj = j; }
    }
    redV[t] = bv; redI[t] = bj;
    __syncthreads();
    for (int s = 128; s > 0; s >>= 1) {
      if (t < s) {
        float v2 = redV[t+s]; int i2 = redI[t+s];
        if (v2 < redV[t] || (v2 == redV[t] && (unsigned)i2 < (unsigned)redI[t])) {
          redV[t] = v2; redI[t] = i2;
        }
      }
      __syncthreads();
    }
    if (t == 0) {
      int e   = bi * NK + k;
      int sel = redI[0];
      Eidx[e]   = sel;
      Dnei[e]   = redV[0];
      IdxOut[e] = (float)sel;
      Drow[sel] = 3.4e38f;     // exclude from next rounds
    }
    __syncthreads();
  }
}

// ---------------------------------------------------------------------------
// Kernel 4 (fused): build 16x192 f16 feature tile in LDS (8 threads/edge) ->
// WMMA GEMM vs Wh[128][192] -> LayerNorm(128) with gamma/beta -> store.
// 128 threads = 4 waves; wave w computes n-tiles n0=w*32 and n0+16.
// ---------------------------------------------------------------------------
__global__ __launch_bounds__(128) void k_edge(
    const float* __restrict__ Ca,   const float* __restrict__ Ofr,
    const int*   __restrict__ Eidx, const float* __restrict__ Dnei,
    const int*   __restrict__ ridx, const int*   __restrict__ chain,
    const float* __restrict__ Wpos, const float* __restrict__ bpos,
    const _Float16* __restrict__ Wh,
    const float* __restrict__ gamma, const float* __restrict__ beta,
    float* __restrict__ Eout)
{
  __shared__ alignas(16) _Float16 Ah[16 * KD];   // 6 KB A-tile (f16)
  __shared__ float Ebuf[16 * NED];               // 8 KB GEMM result
  __shared__ float redS[16 * 8];
  __shared__ float redQ[16 * 8];
  __shared__ float rowMu[16];
  __shared__ float rowRs[16];

  int tid = threadIdx.x;

  // Zero the A-tile (pad columns 167..191 must stay zero).
  {
    unsigned* az = (unsigned*)Ah;
    for (int x = tid; x < 16*KD/2; x += 128) az[x] = 0u;
  }
  __syncthreads();

  // ---- Phase 1: per-edge features, 8 threads per edge (full occupancy) ----
  {
    int le   = tid >> 3;            // edge slot 0..15
    int part = tid & 7;             // work split within edge
    int e    = blockIdx.x * 16 + le;
    int b    = e / (NL*NK);
    int rem  = e - b * (NL*NK);
    int i    = rem / NK;
    int j    = Eidx[e];
    const float* CaB = Ca + (size_t)b * NL * 3;

    float Ci[3], Pi[3] = {0,0,0}, Ni[3] = {0,0,0};
    float Cj[3], Pj[3] = {0,0,0}, Nj[3] = {0,0,0};
    #pragma unroll
    for (int c = 0; c < 3; ++c) { Ci[c] = CaB[i*3+c]; Cj[c] = CaB[j*3+c]; }
    if (i > 0)      { for (int c = 0; c < 3; ++c) Pi[c] = CaB[(i-1)*3+c]; }
    if (i < NL-1)   { for (int c = 0; c < 3; ++c) Ni[c] = CaB[(i+1)*3+c]; }
    if (j > 0)      { for (int c = 0; c < 3; ++c) Pj[c] = CaB[(j-1)*3+c]; }
    if (j < NL-1)   { for (int c = 0; c < 3; ++c) Nj[c] = CaB[(j+1)*3+c]; }

    float d[9];
    d[0] = Dnei[e];
    {
      const float* A[8] = { Pi, Ni, Pi, Pi, Ci, Ci, Ni, Ni };
      const float* Bm[8]= { Pj, Nj, Cj, Nj, Pj, Nj, Pj, Cj };
      #pragma unroll
      for (int g = 0; g < 8; ++g) {
        float dx = A[g][0]-Bm[g][0], dy = A[g][1]-Bm[g][1], dz = A[g][2]-Bm[g][2];
        d[g+1] = sqrtf(dx*dx + dy*dy + dz*dz + 1e-6f);
      }
    }

    _Float16* row = &Ah[le * KD];

    // This part's 18 of the 144 RBF values (9 groups x 16, sigma = 1.25)
    #pragma unroll
    for (int q = 0; q < 18; ++q) {
      int v = part*18 + q;
      int g = v >> 4;
      int m = v & 15;
      float mu = 2.0f + (20.0f/15.0f) * (float)m;
      float z  = (d[g] - mu) * 0.8f;          // 1/1.25
      row[16 + v] = (_Float16)__expf(-z*z);
    }

    if (part == 0) {
      // Positional embedding lookup (16 feats)
      int off = ridx[b*NL + i] - ridx[b*NL + j];
      int ech = (chain[b*NL + i] == chain[b*NL + j]);
      int dd  = off + 32; dd = dd < 0 ? 0 : (dd > 64 ? 64 : dd);
      int dcl = ech ? dd : 65;
      #pragma unroll
      for (int f = 0; f < 16; ++f)
        row[f] = (_Float16)(Wpos[f*66 + dcl] + bpos[f]);
    }

    if (part == 1) {
      // Orientation features: dU (3) + quaternion (4)
      const float* Oi = Ofr + (size_t)(b*NL + i) * 9;
      const float* Oj = Ofr + (size_t)(b*NL + j) * 9;
      float dX[3] = { Cj[0]-Ci[0], Cj[1]-Ci[1], Cj[2]-Ci[2] };
      float tv[3];
      #pragma unroll
      for (int s = 0; s < 3; ++s)
        tv[s] = Oi[s*3+0]*dX[0] + Oi[s*3+1]*dX[1] + Oi[s*3+2]*dX[2];
      float rn = rsqrtf(dot3(tv,tv) + 1e-12f);
      row[160] = (_Float16)(tv[0]*rn);
      row[161] = (_Float16)(tv[1]*rn);
      row[162] = (_Float16)(tv[2]*rn);

      float R[3][3];
      #pragma unroll
      for (int a = 0; a < 3; ++a)
        #pragma unroll
        for (int m = 0; m < 3; ++m)
          R[a][m] = Oi[0*3+a]*Oj[0*3+m] + Oi[1*3+a]*Oj[1*3+m] + Oi[2*3+a]*Oj[2*3+m];

      float Rxx = R[0][0], Ryy = R[1][1], Rzz = R[2][2];
      float m0 = 0.5f * sqrtf(fabsf(1.f + Rxx - Ryy - Rzz) + 1e-12f);
      float m1 = 0.5f * sqrtf(fabsf(1.f - Rxx + Ryy - Rzz) + 1e-12f);
      float m2 = 0.5f * sqrtf(fabsf(1.f - Rxx - Ryy + Rzz) + 1e-12f);
      float qx = sgnf(R[2][1] - R[1][2]) * m0;
      float qy = sgnf(R[0][2] - R[2][0]) * m1;
      float qz = sgnf(R[1][0] - R[0][1]) * m2;
      float qw = 0.5f * sqrtf(fmaxf(1.f + Rxx + Ryy + Rzz, 0.f) + 1e-12f);
      float qn = rsqrtf(qx*qx + qy*qy + qz*qz + qw*qw + 1e-12f);
      row[163] = (_Float16)(qx*qn);
      row[164] = (_Float16)(qy*qn);
      row[165] = (_Float16)(qz*qn);
      row[166] = (_Float16)(qw*qn);
    }
  }
  __syncthreads();

  // ---- Phase 2: WMMA GEMM: [16 x 192](f16) x [192 x 128](f16) -> f32 ----
  {
    int lane = tid & 31;
    int wave = tid >> 5;
    int r    = lane & 15;     // A-row / B-column within tile
    int half = lane >> 4;     // K half-group selector
    int n0   = wave * 32;

    const _Float16* Wr0 = Wh + (size_t)(n0 + r)      * KD;
    const _Float16* Wr1 = Wh + (size_t)(n0 + 16 + r) * KD;
    __builtin_prefetch(Wr0, 0, 0);   // -> global_prefetch_b8
    __builtin_prefetch(Wr1, 0, 0);

    v8f acc0 = {}; v8f acc1 = {};
    #pragma unroll
    for (int ks = 0; ks < 6; ++ks) {
      HFrag a, b0, b1;
      const _Float16* ap = &Ah[r*KD + ks*32 + half*8];
      a.f[0]  = *(const v4f*)(ap);
      a.f[1]  = *(const v4f*)(ap + 16);
      const _Float16* p0 = Wr0 + ks*32 + half*8;
      b0.f[0] = *(const v4f*)(p0);
      b0.f[1] = *(const v4f*)(p0 + 16);
      const _Float16* p1 = Wr1 + ks*32 + half*8;
      b1.f[0] = *(const v4f*)(p1);
      b1.f[1] = *(const v4f*)(p1 + 16);
      acc0 = __builtin_amdgcn_wmma_f32_16x16x32_f16(false, a.h, false, b0.h,
                                                    (short)0, acc0, false, false);
      acc1 = __builtin_amdgcn_wmma_f32_16x16x32_f16(false, a.h, false, b1.h,
                                                    (short)0, acc1, false, false);
    }
    // Scatter accumulators: lane -> (m = rr + 8*half, n = n0(+16) + r)
    #pragma unroll
    for (int rr = 0; rr < 8; ++rr) {
      int m = rr + 8*half;
      Ebuf[m*NED + n0 + r]      = acc0[rr];
      Ebuf[m*NED + n0 + 16 + r] = acc1[rr];
    }
  }
  __syncthreads();

  // ---- Phase 3: LayerNorm(128) + gamma/beta, store to global ----
  {
    int rr   = tid >> 3;      // row 0..15
    int part = tid & 7;       // 8 partials per row, 16 cols each
    float s1 = 0.f, s2 = 0.f;
    #pragma unroll
    for (int c = part*16; c < part*16 + 16; ++c) {
      float v = Ebuf[rr*NED + c];
      s1 += v; s2 += v*v;
    }
    redS[rr*8 + part] = s1;
    redQ[rr*8 + part] = s2;
    __syncthreads();
    if (part == 0) {
      float S1 = 0.f, S2 = 0.f;
      #pragma unroll
      for (int p = 0; p < 8; ++p) { S1 += redS[rr*8+p]; S2 += redQ[rr*8+p]; }
      float mean = S1 * (1.f/128.f);
      float var  = S2 * (1.f/128.f) - mean*mean;
      rowMu[rr] = mean;
      rowRs[rr] = rsqrtf(var + 1e-5f);
    }
    __syncthreads();
    float mean = rowMu[rr], rstd = rowRs[rr];
    size_t e = (size_t)blockIdx.x * 16 + rr;
    float* outp = Eout + e * NED;
    #pragma unroll
    for (int c = part*16; c < part*16 + 16; ++c) {
      float v = (Ebuf[rr*NED + c] - mean) * rstd * gamma[c] + beta[c];
      outp[c] = v;
    }
  }
}

// ---------------------------------------------------------------------------
extern "C" void kernel_launch(void* const* d_in, const int* in_sizes, int n_in,
                              void* d_out, int out_size, void* d_ws, size_t ws_size,
                              hipStream_t stream) {
  const float* Ca    = (const float*)d_in[0];
  const float* mask  = (const float*)d_in[1];
  const int*   ridx  = (const int*)  d_in[2];
  const int*   chain = (const int*)  d_in[3];
  const float* Wpos  = (const float*)d_in[4];
  const float* bpos  = (const float*)d_in[5];
  const float* Wedge = (const float*)d_in[6];
  const float* gamma = (const float*)d_in[7];
  const float* beta  = (const float*)d_in[8];
  float* out = (float*)d_out;

  char* ws = (char*)d_ws;
  float*    Ofr  = (float*)(ws);                         // 4*1536*9*4   = 221184 B
  int*      Eidx = (int*)  (ws + 221184);                // 184320*4     = 737280 B
  float*    Dnei = (float*)(ws + 221184 + 737280);       // 184320*4     = 737280 B
  _Float16* Wh   = (_Float16*)(ws + 221184 + 2*737280);  // 128*192*2    =  49152 B

  // 1) per-node orientation frames
  k_frames<<<(NB*NL + 255)/256, 256, 0, stream>>>(Ca, Ofr);
  // 2) W_edge -> f16 padded, n-major
  k_wcvt<<<(NED*KD + 255)/256, 256, 0, stream>>>(Wedge, Wh);
  // 3) masked distances + top-K (async-LDS staged Ca; writes E_idx tail)
  k_topk<<<NB*NL, 256, 0, stream>>>(Ca, mask, Eidx, Dnei, out + (size_t)NM*NED);
  // 4) fused features -> WMMA GEMM -> LayerNorm -> E output
  k_edge<<<NM/16, 128, 0, stream>>>(Ca, Ofr, Eidx, Dnei, ridx, chain,
                                    Wpos, bpos, Wh, gamma, beta, out);
}
